// SSLMSeed_6760278524417
// MI455X (gfx1250) — matrix operand order repaired and checked by reading, not compiled, not run on hardware
//
#include <hip/hip_runtime.h>
#include <math.h>

// ---------------------------------------------------------------------------
// SSLM recurrent scan for MI455X (gfx1250, wave32, WMMA bf16 16x16x32).
// 4 workgroups x 16 batch rows (batch rows are independent recurrences).
// ---------------------------------------------------------------------------

typedef __bf16 bf16;
typedef bf16  v16bf __attribute__((ext_vector_type(16)));
typedef bf16  v8bf  __attribute__((ext_vector_type(8)));
typedef float v8f   __attribute__((ext_vector_type(8)));

#define VOCAB   256
#define EMB     128
#define HID     256
#define SLOTS   16
#define BATCH   64
#define BPW     16          // batch rows per workgroup (== WMMA M)
#define NWG     (BATCH/BPW) // 4 workgroups
#define TSEQ    2048
#define STEPS   2047
#define THREADS 512         // 16 wave32s
#define BETA_C  0.02f
#define THRESH_C 5.0f

// workspace layout (bytes)
// bf16 weights, packed:
//   emb   [256][128]  @ elt 0
//   x2h   [768][128]  @ elt 32768
//   h2h   [768][256]  @ elt 131072
//   key   [256][256]  @ elt 327680
//   val   [256][256]  @ elt 393216
//   out   [256][256]  @ elt 458752   (total 524288 bf16 = 1 MiB)
static const size_t ELT_EMB = 0;
static const size_t ELT_X2H = 32768;
static const size_t ELT_H2H = 131072;
static const size_t ELT_KEY = 327680;
static const size_t ELT_VAL = 393216;
static const size_t ELT_OUT = 458752;
static const size_t N_BF    = 524288;
static const size_t OFF_MK  = N_BF * 2;                    // f32 mk [64][16][256]
static const size_t OFF_MV  = OFF_MK + 262144ull * 4;      // f32 mv [64][16][256]
static const size_t OFF_SUM = OFF_MV + 262144ull * 4;      // f32 partial[NWG]

// ---------------- WMMA helpers (layouts per CDNA5 ISA) ----------

__device__ __forceinline__ v8f wmma_bf16(v16bf a, v16bf b, v8f c) {
    // D(f32 16x16) = A(16x32 bf16) * B(32x16 bf16) + C
    return __builtin_amdgcn_wmma_f32_16x16x32_bf16(
        /*neg_a=*/false, a, /*neg_b=*/false, b,
        /*c_mod=*/(short)0, c, /*reuse_a=*/false, /*reuse_b=*/false);
}

// A-matrix tile (16 rows x 32 K) from LDS row-major [16][K] bf16.
// lane l<16 : M=l, K-pairs {0..7} in v0-3, {16..23} in v4-7
// lane l>=16: M=l-16, K-pairs {8..15} in v0-3, {24..31} in v4-7
__device__ __forceinline__ v16bf load_a(const bf16* base, int K, int kt, int lane) {
    int row = lane & 15, hi = lane >> 4;
    const bf16* p = base + row * K + kt * 32 + hi * 8;
    v8bf lo = *(const v8bf*)(p);
    v8bf up = *(const v8bf*)(p + 16);
    return __builtin_shufflevector(lo, up, 0,1,2,3,4,5,6,7,8,9,10,11,12,13,14,15);
}

// B-matrix tile (32 K x 16 N) where B = W^T and W is row-major [N_total][K] bf16.
// lane l<16 : N=l,    K = kt*32 + {0..7, 16..23}
// lane l>=16: N=l-16, K = kt*32 + {8..15, 24..31}
__device__ __forceinline__ v16bf load_b(const bf16* W, int K, int nrow0, int kt, int lane) {
    int col = lane & 15, hi = lane >> 4;
    const bf16* p = W + (size_t)(nrow0 + col) * K + kt * 32 + hi * 8;
    v8bf lo = *(const v8bf*)(p);
    v8bf up = *(const v8bf*)(p + 16);
    return __builtin_shufflevector(lo, up, 0,1,2,3,4,5,6,7,8,9,10,11,12,13,14,15);
}

// ---------------- wave32 reductions ----------------------------------------

__device__ __forceinline__ float wave_sum(float v) {
    #pragma unroll
    for (int o = 16; o > 0; o >>= 1) v += __shfl_xor(v, o, 32);
    return v;
}
__device__ __forceinline__ float wave_max(float v) {
    #pragma unroll
    for (int o = 16; o > 0; o >>= 1) v = fmaxf(v, __shfl_xor(v, o, 32));
    return v;
}
__device__ __forceinline__ float sigmoidf_(float x) { return 1.f / (1.f + __expf(-x)); }

// ---------------- init: f32 weights -> bf16 ws, zero mk/mv -----------------

__global__ void sslm_init(const float* __restrict__ emb, const float* __restrict__ x2h,
                          const float* __restrict__ h2h, const float* __restrict__ keyw,
                          const float* __restrict__ valw, const float* __restrict__ outw,
                          char* __restrict__ ws) {
    size_t i = (size_t)blockIdx.x * blockDim.x + threadIdx.x;
    bf16* wsbf = (bf16*)ws;
    if (i < N_BF) {
        float v;
        if      (i < ELT_X2H) v = emb [i - ELT_EMB];
        else if (i < ELT_H2H) v = x2h [i - ELT_X2H];
        else if (i < ELT_KEY) v = h2h [i - ELT_H2H];
        else if (i < ELT_VAL) v = keyw[i - ELT_KEY];
        else if (i < ELT_OUT) v = valw[i - ELT_VAL];
        else                  v = outw[i - ELT_OUT];
        wsbf[i] = (bf16)v;
        ((float*)(ws + OFF_MK))[i] = 0.f;   // zero mk+mv (contiguous 524288 f32)
    }
    if (i < NWG) ((float*)(ws + OFF_SUM))[i] = 0.f;
}

// ---------------- recurrent kernel: one workgroup per 16 batch rows --------

__global__ __launch_bounds__(THREADS)
void sslm_recurrent(const int* __restrict__ tokens,
                    const float* __restrict__ x2h_b,
                    const float* __restrict__ out_b,
                    const float* __restrict__ gate_w,
                    const float* __restrict__ gate_b,
                    char* __restrict__ ws) {
    const bf16* wsbf   = (const bf16*)ws;
    const bf16* emb_bf = wsbf + ELT_EMB;
    const bf16* x2h_bf = wsbf + ELT_X2H;
    const bf16* h2h_bf = wsbf + ELT_H2H;
    const bf16* key_bf = wsbf + ELT_KEY;
    const bf16* val_bf = wsbf + ELT_VAL;
    const bf16* out_bf = wsbf + ELT_OUT;
    float* mkg  = (float*)(ws + OFF_MK);
    float* mvg  = (float*)(ws + OFF_MV);
    float* psum = (float*)(ws + OFF_SUM);

    __shared__ __align__(32) float h32 [BPW * HID];   // f32 hidden carry   16 KB
    __shared__ __align__(32) bf16  hbf [BPW * HID];   // bf16 h (A matrix)   8 KB
    __shared__ __align__(32) bf16  xbf [BPW * EMB];   // bf16 embeddings     4 KB
    __shared__ __align__(32) bf16  sAbf[BPW * HID];   // q_raw then logits   8 KB
    __shared__ __align__(32) bf16  qnbf[BPW * HID];   // q_norm == k_new     8 KB
    __shared__ __align__(32) bf16  vbf [BPW * HID];   // v_new               8 KB
    __shared__ __align__(32) bf16  hmbf[BPW * HID];   // h + mem             8 KB
    __shared__ float attL[BPW * SLOTS];               // att -> softmax w    1 KB
    __shared__ float ageL[BPW * SLOTS];               // slot ages           1 KB
    __shared__ float emaL[BPW], rqL[BPW], nllrow[BPW], gateL[BPW], nllsum[BPW];
    __shared__ int   slotL[BPW];

    const int tid   = threadIdx.x;
    const int lane  = tid & 31;
    const int wave  = tid >> 5;            // 0..15
    const int bbase = blockIdx.x * BPW;
    const int hi    = lane >> 4;
    const int nlo   = lane & 15;

    for (int i = tid; i < BPW * HID; i += THREADS) { h32[i] = 0.f; hbf[i] = (bf16)0.f; }
    for (int i = tid; i < BPW * SLOTS; i += THREADS) ageL[i] = 0.f;
    if (tid < BPW) { emaL[tid] = THRESH_C; nllsum[tid] = 0.f; }
    __syncthreads();

    const float gw0 = gate_w[0], gw1 = gate_w[1], gb = gate_b[0];

    for (int t = 0; t < STEPS; ++t) {
        // ---- stage embeddings for this step ----
        for (int i = tid; i < BPW * EMB; i += THREADS) {
            int b = i >> 7, k = i & (EMB - 1);
            int tok = tokens[(size_t)(bbase + b) * TSEQ + t];
            xbf[i] = emb_bf[tok * EMB + k];
        }
        __syncthreads();

        // ---- GRU gates: wave w owns hidden-column chunk c = w (16 cols) ----
        {
            const int c = wave;
            v8f az = {}, ar = {}, axn = {}, ahn = {};
            #pragma unroll
            for (int kt = 0; kt < EMB / 32; ++kt) {          // x @ x2h_w.T
                v16bf a = load_a(xbf, EMB, kt, lane);
                az  = wmma_bf16(a, load_b(x2h_bf, EMB, 0 * HID + c * 16, kt, lane), az);
                ar  = wmma_bf16(a, load_b(x2h_bf, EMB, 1 * HID + c * 16, kt, lane), ar);
                axn = wmma_bf16(a, load_b(x2h_bf, EMB, 2 * HID + c * 16, kt, lane), axn);
            }
            #pragma unroll
            for (int kt = 0; kt < HID / 32; ++kt) {          // h @ h2h_w.T
                v16bf a = load_a(hbf, HID, kt, lane);
                az  = wmma_bf16(a, load_b(h2h_bf, HID, 0 * HID + c * 16, kt, lane), az);
                ar  = wmma_bf16(a, load_b(h2h_bf, HID, 1 * HID + c * 16, kt, lane), ar);
                ahn = wmma_bf16(a, load_b(h2h_bf, HID, 2 * HID + c * 16, kt, lane), ahn);
            }
            const int col = c * 16 + nlo;
            const float bz = x2h_b[col], br = x2h_b[HID + col], bn = x2h_b[2 * HID + col];
            #pragma unroll
            for (int v = 0; v < 8; ++v) {
                int row = v + 8 * hi;
                float z = sigmoidf_(az[v] + bz);
                float r = sigmoidf_(ar[v] + br);
                float n = tanhf(axn[v] + bn + r * ahn[v]);
                float ho = h32[row * HID + col];
                h32[row * HID + col] = (1.f - z) * ho + z * n;
            }
        }
        __syncthreads();
        for (int i = tid; i < BPW * HID; i += THREADS) hbf[i] = (bf16)h32[i];
        __syncthreads();

        // ---- q = h@key_w.T, v_new = h@val_w.T : wave w owns N-chunk w ----
        {
            const int n = wave;
            v8f aq = {}, av = {};
            #pragma unroll
            for (int kt = 0; kt < HID / 32; ++kt) {
                v16bf a = load_a(hbf, HID, kt, lane);
                aq = wmma_bf16(a, load_b(key_bf, HID, n * 16, kt, lane), aq);
                av = wmma_bf16(a, load_b(val_bf, HID, n * 16, kt, lane), av);
            }
            const int col = n * 16 + nlo;
            #pragma unroll
            for (int v = 0; v < 8; ++v) {
                int row = v + 8 * hi;
                sAbf[row * HID + col] = (bf16)aq[v];
                vbf [row * HID + col] = (bf16)av[v];
            }
        }
        __syncthreads();

        // ---- normalize q (wave w owns row w) ----
        {
            const int row = wave;
            float ss = 0.f;
            #pragma unroll
            for (int j = 0; j < HID / 32; ++j) {
                float q = (float)sAbf[row * HID + lane + 32 * j];
                ss += q * q;
            }
            ss = wave_sum(ss);
            float sc = 1.f / fmaxf(sqrtf(ss), 1e-12f);
            if (lane == 0) rqL[row] = sc;
            #pragma unroll
            for (int j = 0; j < HID / 32; ++j) {
                int d = lane + 32 * j;
                qnbf[row * HID + d] = (bf16)((float)sAbf[row * HID + d] * sc);
            }
        }
        __syncthreads();

        // ---- attention scores vs normalized memory keys ----
        if (tid < BPW * SLOTS) {
            int b = tid >> 4, s = tid & 15;
            const float* kr = mkg + ((size_t)(bbase + b) * SLOTS + s) * HID;
            float dot = 0.f, kk = 0.f;
            for (int d = 0; d < HID; ++d) {
                float kv = kr[d];
                dot += kv * (float)sAbf[b * HID + d];
                kk  += kv * kv;
            }
            attL[tid] = rqL[b] * dot / fmaxf(sqrtf(kk), 1e-12f);
        }
        __syncthreads();

        // ---- softmax(att*4) over slots, in place ----
        if (tid < BPW) {
            float m = -1e30f;
            #pragma unroll
            for (int s = 0; s < SLOTS; ++s) m = fmaxf(m, attL[tid * SLOTS + s]);
            float e[SLOTS]; float sum = 0.f;
            #pragma unroll
            for (int s = 0; s < SLOTS; ++s) { e[s] = __expf(4.f * (attL[tid * SLOTS + s] - m)); sum += e[s]; }
            float inv = 1.f / sum;
            #pragma unroll
            for (int s = 0; s < SLOTS; ++s) attL[tid * SLOTS + s] = e[s] * inv;
        }
        __syncthreads();

        // ---- hm = h + sum_s w[s]*mv[s] ----
        for (int i = tid; i < BPW * HID; i += THREADS) {
            int b = i >> 8, d = i & (HID - 1);
            const float* mvb = mvg + ((size_t)(bbase + b) * SLOTS) * HID + d;
            float m = 0.f;
            #pragma unroll
            for (int s = 0; s < SLOTS; ++s) m += attL[b * SLOTS + s] * mvb[s * HID];
            hmbf[i] = (bf16)(h32[i] + m);
        }
        __syncthreads();

        // ---- logits = hm @ out_w.T + out_b (reuse sAbf) ----
        {
            const int n = wave;
            v8f al = {};
            #pragma unroll
            for (int kt = 0; kt < HID / 32; ++kt) {
                v16bf a = load_a(hmbf, HID, kt, lane);
                al = wmma_bf16(a, load_b(out_bf, HID, n * 16, kt, lane), al);
            }
            const int col = n * 16 + nlo;
            const float bo = out_b[col];
            #pragma unroll
            for (int v = 0; v < 8; ++v) {
                int row = v + 8 * hi;
                sAbf[row * HID + col] = (bf16)(al[v] + bo);
            }
        }
        __syncthreads();

        // ---- NLL via stable log-softmax (wave w owns row w) ----
        {
            const int row = wave;
            float m = -1e30f;
            #pragma unroll
            for (int j = 0; j < VOCAB / 32; ++j)
                m = fmaxf(m, (float)sAbf[row * VOCAB + lane + 32 * j]);
            m = wave_max(m);
            float s = 0.f;
            #pragma unroll
            for (int j = 0; j < VOCAB / 32; ++j)
                s += __expf((float)sAbf[row * VOCAB + lane + 32 * j] - m);
            s = wave_sum(s);
            if (lane == 0) {
                int tgt = tokens[(size_t)(bbase + row) * TSEQ + t + 1];
                nllrow[row] = m + __logf(s) - (float)sAbf[row * VOCAB + tgt];
            }
        }
        __syncthreads();

        // ---- gate / ema / slot selection (one thread per batch row) ----
        if (tid < BPW) {
            float nd = nllrow[tid];
            float lp = emaL[tid] - nd;
            emaL[tid] = (1.f - BETA_C) * emaL[tid] + BETA_C * nd;
            float g = sigmoidf_(gw0 * nd + gw1 * fmaxf(lp, 0.f) + gb);
            if (!(nd > THRESH_C)) g = 0.f;
            gateL[tid] = g;
            float best = -1e30f; int bi = 0;
            #pragma unroll
            for (int s = 0; s < SLOTS; ++s) {
                float a1 = ageL[tid * SLOTS + s] + 1.f;
                if (a1 > best) { best = a1; bi = s; }
            }
            #pragma unroll
            for (int s = 0; s < SLOTS; ++s)
                ageL[tid * SLOTS + s] = (s == bi) ? 0.f : (ageL[tid * SLOTS + s] + 1.f);
            slotL[tid] = bi;
            nllsum[tid] += nd;
        }
        __syncthreads();

        // ---- slot memory write: mk/mv[slot] = (1-g)*old + g*new ----
        for (int i = tid; i < 2 * BPW * HID; i += THREADS) {
            int j = i & (BPW * HID - 1);
            int b = j >> 8, d = j & (HID - 1);
            int sl = slotL[b]; float g = gateL[b], kp = 1.f - g;
            size_t gi = ((size_t)(bbase + b) * SLOTS + sl) * HID + d;
            if (i < BPW * HID) mkg[gi] = kp * mkg[gi] + g * (float)qnbf[j];
            else               mvg[gi] = kp * mvg[gi] + g * (float)vbf[j];
        }
        __syncthreads();
    }

    // deterministic per-block partial sum
    if (wave == 0) {
        float v = (lane < BPW) ? nllsum[lane] : 0.f;
        v = wave_sum(v);
        if (lane == 0) psum[blockIdx.x] = v;
    }
}

__global__ void sslm_finalize(const float* __restrict__ psum, float* __restrict__ out) {
    float s = 0.f;
    #pragma unroll
    for (int i = 0; i < NWG; ++i) s += psum[i];
    out[0] = s / (float)((size_t)STEPS * BATCH);
}

// ---------------------------------------------------------------------------

extern "C" void kernel_launch(void* const* d_in, const int* in_sizes, int n_in,
                              void* d_out, int out_size, void* d_ws, size_t ws_size,
                              hipStream_t stream) {
    (void)in_sizes; (void)n_in; (void)out_size; (void)ws_size;
    const int*   tokens = (const int*)  d_in[0];
    const float* emb_w  = (const float*)d_in[1];
    const float* x2h_w  = (const float*)d_in[2];
    const float* x2h_b  = (const float*)d_in[3];
    const float* h2h_w  = (const float*)d_in[4];
    const float* out_w  = (const float*)d_in[5];
    const float* out_b  = (const float*)d_in[6];
    const float* key_w  = (const float*)d_in[7];
    const float* val_w  = (const float*)d_in[8];
    const float* gate_w = (const float*)d_in[9];
    const float* gate_b = (const float*)d_in[10];
    char* ws = (char*)d_ws;

    sslm_init<<<(int)((N_BF + 511) / 512), 512, 0, stream>>>(
        emb_w, x2h_w, h2h_w, key_w, val_w, out_w, ws);
    sslm_recurrent<<<NWG, THREADS, 0, stream>>>(
        tokens, x2h_b, out_b, gate_w, gate_b, ws);
    sslm_finalize<<<1, 1, 0, stream>>>(
        (const float*)(ws + OFF_SUM), (float*)d_out);
}
